// LOP_Adaptive_45397804318767
// MI455X (gfx1250) — matrix operand order repaired and checked by reading, not compiled, not run on hardware
//
#include <hip/hip_runtime.h>
#include <hip/hip_bf16.h>
#include <math.h>

// Problem constants (match reference)
#define BB   2
#define TT   2048
#define DD   1024
#define HH   16
#define HD   64
#define KK   11

typedef __bf16 bhalf;
typedef __attribute__((ext_vector_type(16))) __bf16 bf16x16;
typedef __attribute__((ext_vector_type(8)))  __bf16 bf16x8;
typedef __attribute__((ext_vector_type(8)))  float  f32x8;

__device__ __forceinline__ int lane_id() { return threadIdx.x & 31; }

union ABPack { bf16x16 v; bf16x8 h[2]; };

// A fragment (M=16,K=32), 16-bit layout:
// lanes 0-15  : row m0+l, K = k0..k0+15 contiguous
// lanes 16-31 : row m0+l, K = k0+16..k0+31 contiguous
__device__ __forceinline__ bf16x16 load_a_frag(const bhalf* A, int lda, int m0, int k0) {
  int l = lane_id(); int hf = l >> 4; l &= 15;
  const bhalf* p = A + (size_t)(m0 + l) * lda + k0 + 16 * hf;
  ABPack u;
  u.h[0] = *(const bf16x8*)p;
  u.h[1] = *(const bf16x8*)(p + 8);
  return u.v;
}

// B fragment (K=32,N=16) from a row-major "Bt" where Bt[n][k] = B[k][n]:
// lanes 0-15  : col n0+l, K = k0..k0+15 contiguous
// lanes 16-31 : col n0+l, K = k0+16..k0+31 contiguous
__device__ __forceinline__ bf16x16 load_b_frag(const bhalf* Bt, int ldb, int n0, int k0) {
  int l = lane_id(); int hf = l >> 4; l &= 15;
  return *(const bf16x16*)(Bt + (size_t)(n0 + l) * ldb + k0 + 16 * hf);
}

__device__ __forceinline__ f32x8 wmma_bf16(bf16x16 a, bf16x16 b, f32x8 c) {
  return __builtin_amdgcn_wmma_f32_16x16x32_bf16(false, a, false, b, (short)0, c, false, false);
}

// Async global->LDS copy, 16 bytes per lane (ASYNCcnt-tracked).
__device__ __forceinline__ void async_b128(const bhalf* lds_dst, const bhalf* g) {
  unsigned lo = (unsigned)(uintptr_t)lds_dst;
  unsigned long long ga = (unsigned long long)(uintptr_t)g;
  asm volatile("global_load_async_to_lds_b128 %0, %1, off" :: "v"(lo), "v"(ga) : "memory");
}
__device__ __forceinline__ void wait_async0() {
  asm volatile("s_wait_asynccnt 0x0" ::: "memory");
}

// ---------------- prep kernels ----------------

__global__ void lop_cast_x(const float* __restrict__ x, bhalf* __restrict__ xb, int n) {
  int i = blockIdx.x * blockDim.x + threadIdx.x;
  if (i < n) xb[i] = (bhalf)x[i];
}

// wt[n][k] = Wsel[k][n%1024], n in [0,3072)
__global__ void lop_prep_wqkv(const float* __restrict__ Wq, const float* __restrict__ Wk,
                              const float* __restrict__ Wv, bhalf* __restrict__ wt) {
  int idx = blockIdx.x * blockDim.x + threadIdx.x;
  if (idx >= 3 * DD * DD) return;
  int n = idx >> 10;
  int k = idx & (DD - 1);
  const float* W = (n < DD) ? Wq : ((n < 2 * DD) ? Wk : Wv);
  int c = n & (DD - 1);
  wt[idx] = (bhalf)W[(size_t)k * DD + c];
}

__global__ void lop_prep_wfc(const float* __restrict__ Wf, bhalf* __restrict__ wt) {
  int idx = blockIdx.x * blockDim.x + threadIdx.x;
  if (idx >= DD * DD) return;
  int n = idx >> 10;
  int k = idx & (DD - 1);
  wt[idx] = (bhalf)Wf[(size_t)k * DD + n];
}

// wm[((h*K+dt)*HD+o)*HD+i] = conv_w[h][o][i][dt] * exp(-(|dt-5|-r_h)^2 / 4.5)
__global__ void lop_prep_convw(const float* __restrict__ conv_w, const float* __restrict__ rp,
                               const float* __restrict__ rs, bhalf* __restrict__ wm) {
  int idx = blockIdx.x * blockDim.x + threadIdx.x;
  if (idx >= HH * KK * HD * HD) return;
  int i  = idx & 63;
  int o  = (idx >> 6) & 63;
  int hd_ = idx >> 12;           // h*11 + dt
  int h  = hd_ / KK;
  int dt = hd_ - h * KK;
  float r = fminf(fmaxf(rp[h] * rs[h], 0.f), 5.f);
  float pos = fabsf((float)dt - 5.f);
  float d = pos - r;
  float msk = __expf(-(d * d) / 4.5f);
  wm[idx] = (bhalf)(conv_w[((size_t)(h * HD + o) * HD + i) * KK + dt] * msk);
}

// ------------- shared async tile stager for the dense GEMMs -------------
// Stages A tile (64 rows x 64 k) and B tile (64 rows x 64 k), both row stride DD.
// 512 x 16B chunks per matrix; 128 threads -> 4 chunks each.
__device__ __forceinline__ void gemm_stage64(const bhalf* Ab, const bhalf* Bb,
                                             bhalf* sA, bhalf* sB, int k0, int tid) {
#pragma unroll
  for (int s = 0; s < 4; ++s) {
    int c = tid + s * 128;
    int row = c >> 3, sub = c & 7;          // 16B chunk within 128B row
    async_b128(sA + row * 64 + sub * 8, Ab + (size_t)row * DD + k0 + sub * 8);
    async_b128(sB + row * 64 + sub * 8, Bb + (size_t)row * DD + k0 + sub * 8);
  }
}

// 8 WMMAs on the staged 64x64 tiles: load fragment groups first, then issue WMMAs.
__device__ __forceinline__ void gemm_compute64(const bhalf* A, const bhalf* Bt,
                                               int mrow, f32x8* accs) {
  bf16x16 a0 = load_a_frag(A, 64, mrow, 0);
  bf16x16 a1 = load_a_frag(A, 64, mrow, 32);
  bf16x16 b0 = load_b_frag(Bt, 64, 0,  0);
  bf16x16 b1 = load_b_frag(Bt, 64, 16, 0);
  bf16x16 b2 = load_b_frag(Bt, 64, 32, 0);
  bf16x16 b3 = load_b_frag(Bt, 64, 48, 0);
  accs[0] = wmma_bf16(a0, b0, accs[0]);
  accs[1] = wmma_bf16(a0, b1, accs[1]);
  accs[2] = wmma_bf16(a0, b2, accs[2]);
  accs[3] = wmma_bf16(a0, b3, accs[3]);
  b0 = load_b_frag(Bt, 64, 0,  32);
  b1 = load_b_frag(Bt, 64, 16, 32);
  b2 = load_b_frag(Bt, 64, 32, 32);
  b3 = load_b_frag(Bt, 64, 48, 32);
  accs[0] = wmma_bf16(a1, b0, accs[0]);
  accs[1] = wmma_bf16(a1, b1, accs[1]);
  accs[2] = wmma_bf16(a1, b2, accs[2]);
  accs[3] = wmma_bf16(a1, b3, accs[3]);
}

// ---------------- QKV GEMM (async-LDS staged, bf16 WMMA) ----------------
__global__ __launch_bounds__(128) void lop_gemm_qkv(
    const bhalf* __restrict__ xb, const bhalf* __restrict__ wt,
    const float* __restrict__ bq, const float* __restrict__ bk, const float* __restrict__ bv,
    bhalf* __restrict__ qb, bhalf* __restrict__ kbuf, bhalf* __restrict__ vt) {
  __shared__ __align__(32) bhalf sA[2][64 * 64];
  __shared__ __align__(32) bhalf sB[2][64 * 64];
  int tid = threadIdx.x;
  int wave = tid >> 5;
  int m0 = blockIdx.x * 64 + wave * 16;
  int n0 = blockIdx.y * 64;
  const bhalf* Ab = xb + (size_t)blockIdx.x * 64 * DD;
  const bhalf* Bb = wt + (size_t)n0 * DD;
  f32x8 accs[4] = {{}, {}, {}, {}};

  gemm_stage64(Ab, Bb, sA[0], sB[0], 0, tid);
  int buf = 0;
  for (int k0 = 0; k0 < DD; k0 += 64) {
    wait_async0();
    __syncthreads();
    if (k0 + 64 < DD) gemm_stage64(Ab, Bb, sA[buf ^ 1], sB[buf ^ 1], k0 + 64, tid);
    gemm_compute64(sA[buf], sB[buf], wave * 16, accs);
    buf ^= 1;
  }

  int l = lane_id(); int hf = l >> 4; int cl = l & 15;
#pragma unroll
  for (int j = 0; j < 4; ++j) {
    int n = n0 + j * 16 + cl;
    int which = n >> 10;
    int col = n & (DD - 1);
    int h = col >> 6, hd = col & 63;
    float bias = (which == 0) ? bq[col] : ((which == 1) ? bk[col] : bv[col]);
#pragma unroll
    for (int r = 0; r < 8; ++r) {
      int m = m0 + r + 8 * hf;
      int b = m >> 11;             // T = 2048
      int t = m & (TT - 1);
      float v = accs[j][r] + bias;
      size_t bh = (size_t)(b * HH + h);
      if (which == 0)      qb[(bh * TT + t) * HD + hd] = (bhalf)(v * 0.125f);   // HD^-0.5
      else if (which == 1) kbuf[(bh * TT + t) * HD + hd] = (bhalf)v;
      else                 vt[(bh * HD + hd) * TT + t] = (bhalf)v;
    }
  }
}

// ---------------- flash attention (async-staged 64-key blocks, WMMA, online softmax) ----------------
__global__ __launch_bounds__(128) void lop_attn(
    const bhalf* __restrict__ qb, const bhalf* __restrict__ kb,
    const bhalf* __restrict__ vt, bhalf* __restrict__ aob) {
  __shared__ __align__(32) bhalf sK[2][64 * 64];   // [key][d]
  __shared__ __align__(32) bhalf sV[2][64 * 64];   // [d][key]  (from transposed V)
  __shared__ __align__(32) bhalf pst[4][16 * 64];  // per-wave P staging (C-layout -> A-layout)
  int tid = threadIdx.x;
  int wave = tid >> 5;
  int tile = blockIdx.x * 4 + wave;   // [0, B*H*T/16); all 4 waves share one bh
  int qt = tile & 127;                // T/16 = 128
  int bh = tile >> 7;
  int q0 = qt << 4;
  const bhalf* qp = qb + (size_t)bh * TT * HD;
  const bhalf* kp = kb + (size_t)bh * TT * HD;
  const bhalf* vp = vt + (size_t)bh * HD * TT;

  bf16x16 qa0 = load_a_frag(qp, HD, q0, 0);
  bf16x16 qa1 = load_a_frag(qp, HD, q0, 32);

  f32x8 o[4] = {{}, {}, {}, {}};
  float mrow[8], lrow[8];
#pragma unroll
  for (int r = 0; r < 8; ++r) { mrow[r] = -1e30f; lrow[r] = 0.f; }
  int l = lane_id(); int hf = l >> 4; int cl = l & 15;
  bhalf* myp = pst[wave];

  // cooperative stager: K tile 64x64, V tile 64x64 (512 x 16B chunks each)
  auto stage = [&](bhalf* dK, bhalf* dV, int kb0) {
#pragma unroll
    for (int s = 0; s < 4; ++s) {
      int c = tid + s * 128;
      int row = c >> 3, sub = c & 7;
      async_b128(dK + row * 64 + sub * 8, kp + (size_t)(kb0 + row) * HD + sub * 8);
      async_b128(dV + row * 64 + sub * 8, vp + (size_t)row * TT + kb0 + sub * 8);
    }
  };

  stage(sK[0], sV[0], 0);
  int buf = 0;
  for (int kb0 = 0; kb0 < TT; kb0 += 64) {
    wait_async0();
    __syncthreads();
    if (kb0 + 64 < TT) stage(sK[buf ^ 1], sV[buf ^ 1], kb0 + 64);
    const bhalf* Kt = sK[buf];
    const bhalf* Vt = sV[buf];

    // S block 16x64 (rows=q, cols=keys kb0..kb0+63), q pre-scaled by HD^-0.5
    f32x8 s0 = {}, s1 = {}, s2 = {}, s3 = {};
    {
      bf16x16 t0 = load_b_frag(Kt, 64, 0,  0);
      bf16x16 t1 = load_b_frag(Kt, 64, 16, 0);
      bf16x16 t2 = load_b_frag(Kt, 64, 32, 0);
      bf16x16 t3 = load_b_frag(Kt, 64, 48, 0);
      s0 = wmma_bf16(qa0, t0, s0);
      s1 = wmma_bf16(qa0, t1, s1);
      s2 = wmma_bf16(qa0, t2, s2);
      s3 = wmma_bf16(qa0, t3, s3);
      t0 = load_b_frag(Kt, 64, 0,  32);
      t1 = load_b_frag(Kt, 64, 16, 32);
      t2 = load_b_frag(Kt, 64, 32, 32);
      t3 = load_b_frag(Kt, 64, 48, 32);
      s0 = wmma_bf16(qa1, t0, s0);
      s1 = wmma_bf16(qa1, t1, s1);
      s2 = wmma_bf16(qa1, t2, s2);
      s3 = wmma_bf16(qa1, t3, s3);
    }

    // online softmax per row (row r + 8*hf lives on the 16 lanes of this half)
#pragma unroll
    for (int r = 0; r < 8; ++r) {
      float mx = fmaxf(fmaxf(s0[r], s1[r]), fmaxf(s2[r], s3[r]));
      mx = fmaxf(mx, __shfl_xor(mx, 1, 32));
      mx = fmaxf(mx, __shfl_xor(mx, 2, 32));
      mx = fmaxf(mx, __shfl_xor(mx, 4, 32));
      mx = fmaxf(mx, __shfl_xor(mx, 8, 32));
      float mnew = fmaxf(mrow[r], mx);
      float corr = __expf(mrow[r] - mnew);
      float p0 = __expf(s0[r] - mnew);
      float p1 = __expf(s1[r] - mnew);
      float p2 = __expf(s2[r] - mnew);
      float p3 = __expf(s3[r] - mnew);
      float rsum = (p0 + p1) + (p2 + p3);
      rsum += __shfl_xor(rsum, 1, 32);
      rsum += __shfl_xor(rsum, 2, 32);
      rsum += __shfl_xor(rsum, 4, 32);
      rsum += __shfl_xor(rsum, 8, 32);
      lrow[r] = lrow[r] * corr + rsum;
      mrow[r] = mnew;
#pragma unroll
      for (int j = 0; j < 4; ++j) o[j][r] *= corr;
      s0[r] = p0; s1[r] = p1; s2[r] = p2; s3[r] = p3;
    }

    // stage P (bf16) to LDS row-major [16][64], reload as A fragments
#pragma unroll
    for (int r = 0; r < 8; ++r) {
      myp[(hf * 8 + r) * 64 + 0  + cl] = (bhalf)s0[r];
      myp[(hf * 8 + r) * 64 + 16 + cl] = (bhalf)s1[r];
      myp[(hf * 8 + r) * 64 + 32 + cl] = (bhalf)s2[r];
      myp[(hf * 8 + r) * 64 + 48 + cl] = (bhalf)s3[r];
    }
    asm volatile("s_wait_dscnt 0x0" ::: "memory");  // LDS RAW fence (in-order per wave)
    bf16x16 pa0 = load_a_frag(myp, 64, 0, 0);
    bf16x16 pa1 = load_a_frag(myp, 64, 0, 32);

    // O += P(16x64) @ V(64x64); Vt is [d][key] so cols of V are rows of Vt
    {
      bf16x16 t0 = load_b_frag(Vt, 64, 0,  0);
      bf16x16 t1 = load_b_frag(Vt, 64, 16, 0);
      bf16x16 t2 = load_b_frag(Vt, 64, 32, 0);
      bf16x16 t3 = load_b_frag(Vt, 64, 48, 0);
      o[0] = wmma_bf16(pa0, t0, o[0]);
      o[1] = wmma_bf16(pa0, t1, o[1]);
      o[2] = wmma_bf16(pa0, t2, o[2]);
      o[3] = wmma_bf16(pa0, t3, o[3]);
      t0 = load_b_frag(Vt, 64, 0,  32);
      t1 = load_b_frag(Vt, 64, 16, 32);
      t2 = load_b_frag(Vt, 64, 32, 32);
      t3 = load_b_frag(Vt, 64, 48, 32);
      o[0] = wmma_bf16(pa1, t0, o[0]);
      o[1] = wmma_bf16(pa1, t1, o[1]);
      o[2] = wmma_bf16(pa1, t2, o[2]);
      o[3] = wmma_bf16(pa1, t3, o[3]);
    }
    buf ^= 1;
  }

  // epilogue: normalize, store bf16 [B,H,T,HD]
#pragma unroll
  for (int r = 0; r < 8; ++r) {
    float inv = 1.f / lrow[r];
    size_t rowoff = ((size_t)bh * TT + q0 + hf * 8 + r) * HD;
    aob[rowoff + 0  + cl] = (bhalf)(o[0][r] * inv);
    aob[rowoff + 16 + cl] = (bhalf)(o[1][r] * inv);
    aob[rowoff + 32 + cl] = (bhalf)(o[2][r] * inv);
    aob[rowoff + 48 + cl] = (bhalf)(o[3][r] * inv);
  }
}

// ---------------- grouped temporal conv as WMMA GEMM ----------------
// y[b,h,o,t] = sum_dt sum_i wm[h,dt,o,i] * aob[b,h,clamp(t+dt-5),i] + conv_b[h,o]
// stored to ybf[b][t][h*64+o] (bf16) which is the FC input layout
__global__ __launch_bounds__(128) void lop_conv(
    const bhalf* __restrict__ aob, const bhalf* __restrict__ wm,
    const float* __restrict__ cb, bhalf* __restrict__ ybf) {
  int wave = threadIdx.x >> 5;
  int tt = blockIdx.x & 127;
  int bh = blockIdx.x >> 7;
  int h = bh & (HH - 1);
  int b = bh >> 4;
  int o0 = wave << 4;
  int t0 = tt << 4;
  const bhalf* xin = aob + (size_t)bh * TT * HD;
  int l = lane_id(); int hf = l >> 4; int cl = l & 15;
  f32x8 acc = {};
  for (int dt = 0; dt < KK; ++dt) {
    const bhalf* wA = wm + (size_t)(h * KK + dt) * (HD * HD);
    int ts = t0 + cl + dt - 5;
    ts = ts < 0 ? 0 : (ts > TT - 1 ? TT - 1 : ts);
    const bhalf* xr = xin + (size_t)ts * HD;
    bf16x16 a0 = load_a_frag(wA, HD, o0, 0);
    bf16x16 a1 = load_a_frag(wA, HD, o0, 32);
    bf16x16 b0 = *(const bf16x16*)(xr + 16 * hf);
    bf16x16 b1 = *(const bf16x16*)(xr + 32 + 16 * hf);
    acc = wmma_bf16(a0, b0, acc);
    acc = wmma_bf16(a1, b1, acc);
  }
#pragma unroll
  for (int r = 0; r < 8; ++r) {
    int o = o0 + hf * 8 + r;
    float v = acc[r] + cb[h * HD + o];
    ybf[((size_t)(b * TT + t0 + cl)) * DD + h * HD + o] = (bhalf)v;
  }
}

// ---------------- FC GEMM (async-LDS staged, bf16 WMMA -> fp32) ----------------
__global__ __launch_bounds__(128) void lop_gemm_fc(
    const bhalf* __restrict__ ybf, const bhalf* __restrict__ wt,
    const float* __restrict__ fb, float* __restrict__ outp) {
  __shared__ __align__(32) bhalf sA[2][64 * 64];
  __shared__ __align__(32) bhalf sB[2][64 * 64];
  int tid = threadIdx.x;
  int wave = tid >> 5;
  int m0 = blockIdx.x * 64 + wave * 16;
  int n0 = blockIdx.y * 64;
  const bhalf* Ab = ybf + (size_t)blockIdx.x * 64 * DD;
  const bhalf* Bb = wt + (size_t)n0 * DD;
  f32x8 accs[4] = {{}, {}, {}, {}};

  gemm_stage64(Ab, Bb, sA[0], sB[0], 0, tid);
  int buf = 0;
  for (int k0 = 0; k0 < DD; k0 += 64) {
    wait_async0();
    __syncthreads();
    if (k0 + 64 < DD) gemm_stage64(Ab, Bb, sA[buf ^ 1], sB[buf ^ 1], k0 + 64, tid);
    gemm_compute64(sA[buf], sB[buf], wave * 16, accs);
    buf ^= 1;
  }

  int l = lane_id(); int hf = l >> 4; int cl = l & 15;
#pragma unroll
  for (int j = 0; j < 4; ++j) {
    int n = n0 + j * 16 + cl;
    float bias = fb[n];
#pragma unroll
    for (int r = 0; r < 8; ++r)
      outp[(size_t)(m0 + r + 8 * hf) * DD + n] = accs[j][r] + bias;
  }
}

// ---------------- residual + LayerNorm ----------------
__global__ __launch_bounds__(256) void lop_resid_ln(
    const float* __restrict__ fcout, const float* __restrict__ x,
    const float* __restrict__ g, const float* __restrict__ bta, float* __restrict__ out) {
  __shared__ float ssum[8], ssq[8];
  int row = blockIdx.x;
  int tid = threadIdx.x;
  const float* fr = fcout + (size_t)row * DD;
  const float* xr = x + (size_t)row * DD;
  float vals[4];
  float lsum = 0.f, lsq = 0.f;
#pragma unroll
  for (int j = 0; j < 4; ++j) {
    int c = tid + j * 256;
    float v = fr[c] + xr[c];
    vals[j] = v;
    lsum += v;
    lsq += v * v;
  }
#pragma unroll
  for (int ofs = 16; ofs >= 1; ofs >>= 1) {
    lsum += __shfl_xor(lsum, ofs, 32);
    lsq  += __shfl_xor(lsq,  ofs, 32);
  }
  if ((tid & 31) == 0) { ssum[tid >> 5] = lsum; ssq[tid >> 5] = lsq; }
  __syncthreads();
  float tot = 0.f, totq = 0.f;
#pragma unroll
  for (int w = 0; w < 8; ++w) { tot += ssum[w]; totq += ssq[w]; }
  float mean = tot * (1.f / DD);
  float var = totq * (1.f / DD) - mean * mean;
  float rstd = rsqrtf(var + 1e-5f);
#pragma unroll
  for (int j = 0; j < 4; ++j) {
    int c = tid + j * 256;
    out[(size_t)row * DD + c] = (vals[j] - mean) * rstd * g[c] + bta[c];
  }
}

// ---------------- launch ----------------
extern "C" void kernel_launch(void* const* d_in, const int* in_sizes, int n_in,
                              void* d_out, int out_size, void* d_ws, size_t ws_size,
                              hipStream_t stream) {
  const float* x   = (const float*)d_in[0];
  const float* Wq  = (const float*)d_in[1];
  const float* bq  = (const float*)d_in[2];
  const float* Wk  = (const float*)d_in[3];
  const float* bk  = (const float*)d_in[4];
  const float* Wv  = (const float*)d_in[5];
  const float* bv  = (const float*)d_in[6];
  const float* rp  = (const float*)d_in[7];
  const float* rs  = (const float*)d_in[8];
  const float* cw  = (const float*)d_in[9];
  const float* cbv = (const float*)d_in[10];
  const float* Wf  = (const float*)d_in[11];
  const float* fb  = (const float*)d_in[12];
  const float* lg  = (const float*)d_in[13];
  const float* lb  = (const float*)d_in[14];
  float* out = (float*)d_out;

  char* ws = (char*)d_ws;
  size_t off = 0;
  auto take = [&](size_t bytes) -> char* {
    char* p = ws + off;
    off = (off + bytes + 255) & ~(size_t)255;
    return p;
  };
  const size_t NTOK = (size_t)BB * TT;           // 4096
  bhalf* xb    = (bhalf*)take(NTOK * DD * 2);
  bhalf* wqkvt = (bhalf*)take((size_t)3 * DD * DD * 2);
  bhalf* wfct  = (bhalf*)take((size_t)DD * DD * 2);
  bhalf* wmask = (bhalf*)take((size_t)HH * KK * HD * HD * 2);
  bhalf* qb    = (bhalf*)take(NTOK * DD * 2);
  bhalf* kbuf  = (bhalf*)take(NTOK * DD * 2);
  bhalf* vt    = (bhalf*)take(NTOK * DD * 2);
  bhalf* aob   = (bhalf*)take(NTOK * DD * 2);
  bhalf* ybf   = (bhalf*)take(NTOK * DD * 2);
  float* fcout = (float*)take(NTOK * DD * 4);
  (void)in_sizes; (void)n_in; (void)out_size; (void)ws_size;

  lop_cast_x<<<(NTOK * DD) / 256, 256, 0, stream>>>(x, xb, (int)(NTOK * DD));
  lop_prep_wqkv<<<(3 * DD * DD) / 256, 256, 0, stream>>>(Wq, Wk, Wv, wqkvt);
  lop_prep_wfc<<<(DD * DD) / 256, 256, 0, stream>>>(Wf, wfct);
  lop_prep_convw<<<(HH * KK * HD * HD) / 256, 256, 0, stream>>>(cw, rp, rs, wmask);

  lop_gemm_qkv<<<dim3(64, 48), 128, 0, stream>>>(xb, wqkvt, bq, bk, bv, qb, kbuf, vt);
  lop_attn<<<1024, 128, 0, stream>>>(qb, kbuf, vt, aob);
  lop_conv<<<4096, 128, 0, stream>>>(aob, wmask, cbv, ybf);
  lop_gemm_fc<<<dim3(64, 16), 128, 0, stream>>>(ybf, wfct, fb, fcout);
  lop_resid_ln<<<4096, 256, 0, stream>>>(fcout, x, lg, lb, out);
}
